// HierarchicalAttention_57140244906448
// MI455X (gfx1250) — matrix operand order repaired and checked by reading, not compiled, not run on hardware
//
#include <hip/hip_runtime.h>
#include <hip/hip_bf16.h>

// CDNA5 / gfx1250 hierarchical attention (word -> sentence -> section pooling).
// Compute-bound (~86 GFLOP vs 134 MB input) => all GEMMs via v_wmma_f32_16x16x32_f16.
// Weights are pre-converted to f16 in workspace (L2-resident); each block stages
// its x-slice into LDS as f16 once, so inner GEMM loops are pure
// ds_load_b128 / global_load_b128 + v_wmma with no per-iteration cvt chains.
// V is stored transposed in LDS so every B-fragment load is contiguous.
// Masked mean-pool fused into the out-projection via ds_add_f32.

typedef __attribute__((ext_vector_type(16))) _Float16 v16h;
typedef __attribute__((ext_vector_type(8)))  _Float16 v8h;
typedef __attribute__((ext_vector_type(8)))  float    v8f;
typedef __attribute__((ext_vector_type(4)))  float    v4f;

#define DEVINL __device__ __forceinline__

// ---- WMMA fragment loaders (layouts per CDNA5 ISA 7.12.2, wave32) ------------
// A (16x32 f16): lane 0-15 holds row M=lane, K=0..7 then 16..23;
//                lane 16-31 holds row M=lane-16, K=8..15 then 24..31.
DEVINL v16h frag_a_h(const _Float16* base, int stride, int lane) {
  int row = lane & 15;
  int kb  = (lane >> 4) << 3;                 // 0 or 8
  const _Float16* p = base + row * stride + kb;
  v16h a;
  *((v8h*)(&a) + 0) = *(const v8h*)(p);       // K = kb .. kb+7
  *((v8h*)(&a) + 1) = *(const v8h*)(p + 16);  // K = kb+16 .. kb+23
  return a;
}

// B (32x16 f16) read from a transposed (N-major) buffer Bt[n][k]:
// lane 0-15: col N=lane, K=0..7 then 16..23 ; lane 16-31: col N=lane-16, K=8..15 then 24..31.
DEVINL v16h frag_b_h(const _Float16* base, int stride, int lane) {
  int col = lane & 15;
  int kb  = (lane >> 4) << 3;                 // 0 or 8
  const _Float16* p = base + col * stride + kb;
  v16h b;
  *((v8h*)(&b) + 0) = *(const v8h*)(p);
  *((v8h*)(&b) + 1) = *(const v8h*)(p + 16);
  return b;
}

DEVINL v8f wmma_f16(v16h a, v16h b, v8f c) {
  // (neg_a, A, neg_b, B, c_mod, C, reuse_a, reuse_b)
  return __builtin_amdgcn_wmma_f32_16x16x32_f16(false, a, false, b, (short)0, c,
                                                false, false);
}

// -----------------------------------------------------------------------------
// One MHA + masked mean pool level. One block per sequence, 256 threads (8 waves).
//   x:      [nseq, Lpad, 256] f32 (pad rows zero)
//   mask:   [nseq, Lpad] f32 in {0,1} (pad rows zero)
//   Wqkv16: [768,256] f16 (N-major == B^T), Wo16: [256,256] f16
//   pooled: written at (seq/grp)*grpStride + (seq%grp)*256
// -----------------------------------------------------------------------------
__global__ __launch_bounds__(256)
void mha_pool_kernel(const float* __restrict__ x, const float* __restrict__ mask,
                     const _Float16* __restrict__ Wqkv16, const float* __restrict__ bqkv,
                     const _Float16* __restrict__ Wo16, const float* __restrict__ bo,
                     float* __restrict__ pooled, int Lpad, int grp, int grpStride) {
  constexpr int H = 256, NH = 4, DH = 64;

  // sXO: x-staging (f16, stride 264) during stage 1; reused as attention output
  // sO (f16, stride 256) afterwards -- lifetimes are disjoint.
  __shared__ _Float16 sXO[64 * 264];
  __shared__ _Float16 sQK[64 * 520];   // Q cols 0..255, K cols 256..511 (row-major)
  __shared__ _Float16 sVt[256 * 64];   // V transposed: sVt[d][l]
  __shared__ float    sS [64 * 68];    // scores, one head at a time (f32)
  __shared__ _Float16 sP [64 * 64];    // softmax probs, one head (f16)
  __shared__ float    sMask[64];
  __shared__ float    sPool[256];

  const int tid  = threadIdx.x;
  const int lane = tid & 31;
  const int wave = tid >> 5;
  const int seq  = blockIdx.x;
  const int Mt   = Lpad >> 4;               // row tiles (1 or 4)
  const int kA   = (Lpad + 31) >> 5;        // K-steps over the L dimension

  if (tid < 64) sMask[tid] = (tid < Lpad) ? mask[(size_t)seq * Lpad + tid] : 0.0f;
  if (Lpad < 32) {                          // zero the K-padding read by WMMA
    for (int i = tid; i < 256 * 64; i += 256) sVt[i] = (_Float16)0.0f;
    for (int i = tid; i < 64 * 64; i += 256)  sP[i]  = (_Float16)0.0f;
  }

  // ---- Stage 0: stage x slice into LDS as f16 (one convert per element) ----
  const float* xs = x + (size_t)seq * Lpad * H;
  for (int i = tid; i < Lpad * 64; i += 256) {
    int row = i >> 6, c4 = (i & 63) << 2;
    v4f f = *(const v4f*)(xs + row * H + c4);
    _Float16* d = sXO + row * 264 + c4;
    d[0] = (_Float16)f[0];
    d[1] = (_Float16)f[1];
    d[2] = (_Float16)f[2];
    d[3] = (_Float16)f[3];
  }
  __syncthreads();

  // ---- Stage 1: QKV = x @ Wqkv^T + bqkv  (M=Lpad, N=768, K=256) ----
  for (int t = wave; t < Mt * 48; t += 8) {
    int mt = t / 48, nt = t % 48;
    v8f acc = {};
    const _Float16* ab = sXO + mt * 16 * 264;
    const _Float16* bb = Wqkv16 + nt * 16 * H;   // Wqkv[n][k] == (Wqkv^T)[k][n]
#pragma unroll
    for (int ks = 0; ks < 8; ++ks) {
      v16h a = frag_a_h(ab + ks * 32, 264, lane);
      v16h b = frag_b_h(bb + ks * 32, H, lane);
      acc = wmma_f16(a, b, acc);
    }
    int col = nt * 16 + (lane & 15);
    int rb  = mt * 16 + ((lane >> 4) << 3);
    float bias = bqkv[col];
    if (nt < 32) {                            // Q or K: row-major
#pragma unroll
      for (int r = 0; r < 8; ++r)
        sQK[(rb + r) * 520 + col] = (_Float16)(acc[r] + bias);
    } else {                                  // V: store transposed
      int d = col - 512;
#pragma unroll
      for (int r = 0; r < 8; ++r)
        sVt[d * 64 + rb + r] = (_Float16)(acc[r] + bias);
    }
  }
  __syncthreads();

  // ---- Stage 2: per-head attention (sXO now dead; reuse it as sO) ----
  _Float16* sO = sXO;                          // stride 256 from here on
  for (int h = 0; h < NH; ++h) {
    // scores = Q_h @ K_h^T * 0.125 + key-pad bias   (M=N=Lpad, K=64)
    for (int t = wave; t < Mt * Mt; t += 8) {
      int mt = t / Mt, nt = t % Mt;
      v8f acc = {};
      const _Float16* ab = &sQK[mt * 16 * 520 + h * DH];        // Q rows
      const _Float16* bb = &sQK[nt * 16 * 520 + 256 + h * DH];  // K rows as B^T
#pragma unroll
      for (int ks = 0; ks < 2; ++ks) {
        v16h a = frag_a_h(ab + ks * 32, 520, lane);
        v16h b = frag_b_h(bb + ks * 32, 520, lane);
        acc = wmma_f16(a, b, acc);
      }
      int col = nt * 16 + (lane & 15);
      int rb  = mt * 16 + ((lane >> 4) << 3);
      float mb = (1.0f - sMask[col]) * -1e9f;
#pragma unroll
      for (int r = 0; r < 8; ++r)
        sS[(rb + r) * 68 + col] = acc[r] * 0.125f + mb;
    }
    __syncthreads();

    // row softmax -> sP (f16, zero-padded to kA*32 cols)
    for (int row = tid; row < Lpad; row += 256) {
      float m = -3.4e38f;
      for (int j = 0; j < Lpad; ++j) m = fmaxf(m, sS[row * 68 + j]);
      float s = 0.0f;
      for (int j = 0; j < Lpad; ++j) s += __expf(sS[row * 68 + j] - m);
      float inv = 1.0f / s;
      int kcols = kA * 32;
      for (int j = 0; j < kcols; ++j)
        sP[row * 64 + j] =
            (_Float16)((j < Lpad) ? __expf(sS[row * 68 + j] - m) * inv : 0.0f);
    }
    __syncthreads();

    // O_h = P @ V_h   (M=Lpad, N=64, K=Lpad)  -- B from transposed V (contiguous)
    for (int t = wave; t < Mt * 4; t += 8) {
      int mt = t >> 2, nt = t & 3;
      v8f acc = {};
      const _Float16* ab = &sP[mt * 16 * 64];
      const _Float16* bb = &sVt[(h * DH + nt * 16) * 64];
      for (int ks = 0; ks < kA; ++ks) {
        v16h a = frag_a_h(ab + ks * 32, 64, lane);
        v16h b = frag_b_h(bb + ks * 32, 64, lane);
        acc = wmma_f16(a, b, acc);
      }
      int col = h * DH + nt * 16 + (lane & 15);
      int rb  = mt * 16 + ((lane >> 4) << 3);
#pragma unroll
      for (int r = 0; r < 8; ++r)
        sO[(rb + r) * 256 + col] = (_Float16)acc[r];
    }
    __syncthreads();
  }

  // ---- Stage 3: out = O @ Wo^T + bo, fused masked mean pool ----
  sPool[tid] = 0.0f;
  __syncthreads();
  for (int t = wave; t < Mt * 16; t += 8) {
    int mt = t >> 4, nt = t & 15;
    v8f acc = {};
    const _Float16* ab = sO + mt * 16 * 256;
    const _Float16* bb = Wo16 + nt * 16 * H;   // Wo[n][k]
#pragma unroll
    for (int ks = 0; ks < 8; ++ks) {
      v16h a = frag_a_h(ab + ks * 32, 256, lane);
      v16h b = frag_b_h(bb + ks * 32, H, lane);
      acc = wmma_f16(a, b, acc);
    }
    int col = nt * 16 + (lane & 15);
    int rb  = mt * 16 + ((lane >> 4) << 3);
    float bias = bo[col];
#pragma unroll
    for (int r = 0; r < 8; ++r)
      atomicAdd(&sPool[col], (acc[r] + bias) * sMask[rb + r]);  // ds_add_f32
  }
  __syncthreads();
  {
    float ms = 0.0f;
    for (int l = 0; l < Lpad; ++l) ms += sMask[l];
    float p = sPool[tid] / (ms + 1e-10f);
    pooled[(size_t)(seq / grp) * grpStride + (size_t)(seq % grp) * 256 + tid] = p;
  }
}

// ---------------------------- prep kernels -----------------------------------
__global__ void k_cvt_f16(const float* __restrict__ src, _Float16* __restrict__ dst,
                          int n4) {           // n4 = n/4
  int i = blockIdx.x * blockDim.x + threadIdx.x;
  if (i < n4) {
    v4f f = *(const v4f*)(src + i * 4);
    _Float16* d = dst + i * 4;
    d[0] = (_Float16)f[0];
    d[1] = (_Float16)f[1];
    d[2] = (_Float16)f[2];
    d[3] = (_Float16)f[3];
  }
}

__global__ void k_mask_word(const int* __restrict__ am, float* __restrict__ m1, int n) {
  int i = blockIdx.x * blockDim.x + threadIdx.x;
  if (i < n) m1[i] = (float)am[i];
}

__global__ void k_mask_sent(const int* __restrict__ am, float* __restrict__ m2, int nsent) {
  int i = blockIdx.x * blockDim.x + threadIdx.x;
  if (i < nsent) {
    int s = 0;
    for (int w = 0; w < 64; ++w) s += am[i * 64 + w];
    m2[i] = (s > 0) ? 1.0f : 0.0f;
  }
}

__global__ void k_mask_sect(const int* __restrict__ am, float* __restrict__ m3) {
  int i = blockIdx.x * blockDim.x + threadIdx.x;   // 16*16
  if (i < 256) {
    int b = i >> 4, r = i & 15;
    float v = 0.0f;
    if (r < 8) {
      int s = 0;
      const int* p = am + (size_t)(b * 8 + r) * 16 * 64;
      for (int j = 0; j < 1024; ++j) s += p[j];
      v = (s > 0) ? 1.0f : 0.0f;
    }
    m3[i] = v;
  }
}

__global__ void k_zero_f32(float* __restrict__ p, int n) {
  int i = blockIdx.x * blockDim.x + threadIdx.x;
  if (i < n) p[i] = 0.0f;
}

// ---------------------------- launcher ---------------------------------------
extern "C" void kernel_launch(void* const* d_in, const int* in_sizes, int n_in,
                              void* d_out, int out_size, void* d_ws, size_t ws_size,
                              hipStream_t stream) {
  (void)in_sizes; (void)n_in; (void)out_size; (void)ws_size;

  const float* we  = (const float*)d_in[0];   // [16,8,16,64,256]
  const int*   am  = (const int*)  d_in[1];   // [16,8,16,64]
  const float* wW  = (const float*)d_in[2];   const float* wb  = (const float*)d_in[3];
  const float* wO  = (const float*)d_in[4];   const float* wob = (const float*)d_in[5];
  const float* sW  = (const float*)d_in[6];   const float* sb  = (const float*)d_in[7];
  const float* sO_ = (const float*)d_in[8];   const float* sob = (const float*)d_in[9];
  const float* cW  = (const float*)d_in[10];  const float* cb  = (const float*)d_in[11];
  const float* cO  = (const float*)d_in[12];  const float* cob = (const float*)d_in[13];

  // workspace carve-up: f32 region, then 16B-aligned f16 weight copies
  float* x2 = (float*)d_ws;                 // [128,16,256]  (level-1 pooled)
  float* x3 = x2 + 2048 * 256;              // [16,16,256]   (level-2 pooled, padded)
  float* m1 = x3 + 16 * 16 * 256;           // [2048,64]
  float* m2 = m1 + 2048 * 64;               // [128,16]
  float* m3 = m2 + 2048;                    // [16,16]
  _Float16* wq1h = (_Float16*)(m3 + 256);   // [768,256] f16
  _Float16* wo1h = wq1h + 768 * 256;        // [256,256] f16
  _Float16* wq2h = wo1h + 256 * 256;
  _Float16* wo2h = wq2h + 768 * 256;
  _Float16* wq3h = wo2h + 256 * 256;
  _Float16* wo3h = wq3h + 768 * 256;

  // weight conversion (f32 -> f16, once; L2-resident afterwards)
  const int nq4 = 768 * 256 / 4, no4 = 256 * 256 / 4;
  k_cvt_f16<<<(nq4 + 255) / 256, 256, 0, stream>>>(wW, wq1h, nq4);
  k_cvt_f16<<<(no4 + 255) / 256, 256, 0, stream>>>(wO, wo1h, no4);
  k_cvt_f16<<<(nq4 + 255) / 256, 256, 0, stream>>>(sW, wq2h, nq4);
  k_cvt_f16<<<(no4 + 255) / 256, 256, 0, stream>>>(sO_, wo2h, no4);
  k_cvt_f16<<<(nq4 + 255) / 256, 256, 0, stream>>>(cW, wq3h, nq4);
  k_cvt_f16<<<(no4 + 255) / 256, 256, 0, stream>>>(cO, wo3h, no4);

  // masks + padding
  k_mask_word<<<(2048 * 64 + 255) / 256, 256, 0, stream>>>(am, m1, 2048 * 64);
  k_mask_sent<<<(2048 + 255) / 256, 256, 0, stream>>>(am, m2, 2048);
  k_mask_sect<<<1, 256, 0, stream>>>(am, m3);
  k_zero_f32<<<(16 * 16 * 256 + 255) / 256, 256, 0, stream>>>(x3, 16 * 16 * 256);

  // level 1: words -> sentence embeddings (2048 seqs, L=64)
  mha_pool_kernel<<<2048, 256, 0, stream>>>(we, m1, wq1h, wb, wo1h, wob,
                                            x2, 64, 16, 16 * 256);
  // level 2: sentences -> section embeddings (128 seqs, L=16)
  mha_pool_kernel<<<128, 256, 0, stream>>>(x2, m2, wq2h, sb, wo2h, sob,
                                           x3, 16, 8, 16 * 256);
  // level 3: sections -> doc embeddings (16 seqs, L=8 padded to 16)
  mha_pool_kernel<<<16, 256, 0, stream>>>(x3, m3, wq3h, cb, wo3h, cob,
                                          (float*)d_out, 16, 1, 256);
}
